// SparseRetrieval_4501125726902
// MI455X (gfx1250) — matrix operand order repaired and checked by reading, not compiled, not run on hardware
//
#include <hip/hip_runtime.h>
#include <hip/hip_bf16.h>

// ---------------------------------------------------------------------------
// SparseRetrieval: q/k/v proj -> causal scores -> top-8 sparse softmax ->
// gathered attn*v -> output proj.  All GEMMs via v_wmma_f32_16x16x32_bf16.
// B=4, N=4096, D=1024, K=8.
// GEMM: 32x64 per wave (two row tiles share B fragments); compiler-scheduled
// pipelining (no manual staging -> no spills), immediate-offset addressing.
// Scores: 4 key tiles per pass share one q fragment.
// ---------------------------------------------------------------------------

typedef __attribute__((ext_vector_type(16))) __bf16        v16bf;
typedef __attribute__((ext_vector_type(8)))  __bf16        v8bf;
typedef __attribute__((ext_vector_type(8)))  float         v8f;
typedef __attribute__((ext_vector_type(4)))  unsigned int  v4u;

static constexpr int Bsz  = 4;
static constexpr int Nseq = 4096;
static constexpr int Dm   = 1024;
static constexpr int TOPK = 8;

__device__ __forceinline__ float bfbits_to_f(unsigned int s16) {
    unsigned int u = s16 << 16;
    return __builtin_bit_cast(float, u);
}
__device__ __forceinline__ unsigned short f_to_bfbits(float f) {
    unsigned int u = __builtin_bit_cast(unsigned int, f);
    u += 0x7fffu + ((u >> 16) & 1u);          // round-to-nearest-even
    return (unsigned short)(u >> 16);
}

__device__ __forceinline__ v16bf concat8(v8bf lo, v8bf hi) {
    return __builtin_shufflevector(lo, hi, 0, 1, 2, 3, 4, 5, 6, 7,
                                           8, 9, 10, 11, 12, 13, 14, 15);
}

__device__ __forceinline__ v8f wmma_bf16(v16bf a, v16bf b, v8f c) {
    return __builtin_amdgcn_wmma_f32_16x16x32_bf16(false, a, false, b,
                                                   (short)0, c, false, false);
}

// --------------------- f32 -> bf16 convert (2 at a time) -------------------
__global__ void cvt_f32_to_bf16(const float2* __restrict__ in,
                                unsigned int* __restrict__ out, int npairs) {
    int i = blockIdx.x * blockDim.x + threadIdx.x;
    int stride = gridDim.x * blockDim.x;
    for (; i < npairs; i += stride) {
        float2 f = in[i];
        out[i] = (unsigned int)f_to_bfbits(f.x) |
                 ((unsigned int)f_to_bfbits(f.y) << 16);
    }
}

// ---------------- pack f32 weight into WMMA B-fragment order ---------------
// Layout: [kkblk][ct][lane][r]  (uints), lane-contiguous 32B fragments.
//   out[((kkblk*64 + ct)*32 + lane)*8 + r] = pack(W[k,col], W[k+1,col])
//   k = kkblk*32 + 2r + 16*(lane>>4),  col = ct*16 + (lane&15)
__global__ void pack_weight_bf16(const float* __restrict__ W,
                                 unsigned int* __restrict__ out) {
    const int nt16 = Dm / 16;                 // 64
    const int total = (Dm / 32) * nt16 * 256; // uints
    int o = blockIdx.x * blockDim.x + threadIdx.x;
    int stride = gridDim.x * blockDim.x;
    for (; o < total; o += stride) {
        int r     = o & 7;
        int lane  = (o >> 3) & 31;
        int tile  = o >> 8;           // kkblk*nt16 + ct
        int ct    = tile % nt16;
        int kkblk = tile / nt16;
        int k   = kkblk * 32 + 2 * r + 16 * (lane >> 4);
        int col = ct * 16 + (lane & 15);
        unsigned int lo = f_to_bfbits(W[(size_t)k * Dm + col]);
        unsigned int hi = f_to_bfbits(W[(size_t)(k + 1) * Dm + col]);
        out[o] = lo | (hi << 16);
    }
}

// ------------------------------- WMMA GEMM ---------------------------------
// C[M,Dm] = A[M,Dm] * Bpacked + bias.  One wave computes a 32x64 strip: two
// 16-row tiles share the four B fragments (8 WMMAs per k-step).
// blockDim = 128 (4 waves), grid = (Dm/256, M/32).  K/N hardcoded to Dm so
// all secondary offsets are immediates (fewer live pointers -> no spills).
__global__ void wmma_gemm_bf16(const __bf16* __restrict__ A,
                               const v16bf* __restrict__ Wp,  // packed fragments
                               const float* __restrict__ bias,
                               unsigned short* __restrict__ outBf,  // bf16 out (or null)
                               float* __restrict__ outF) {          // f32 out (or null)
    const int lane = threadIdx.x & 31;
    const int wave = threadIdx.x >> 5;
    const int half = lane >> 4;
    const int l15  = lane & 15;
    const int rowBase = blockIdx.y * 32;
    const int colBase = (blockIdx.x * 4 + wave) * 64;
    constexpr int nt16  = Dm / 16;            // 64
    constexpr int bStep = nt16 * 32;          // v16bf per k-step
    constexpr int aOff1 = 2 * Dm;             // +16 rows, in v8bf units (32KB)

    v8f acc0[4] = {v8f{}, v8f{}, v8f{}, v8f{}};
    v8f acc1[4] = {v8f{}, v8f{}, v8f{}, v8f{}};

    const v8bf*  aPtr = (const v8bf*)(A + (size_t)(rowBase + l15) * Dm + 16 * half);
    const v16bf* bPtr = Wp + (size_t)(colBase >> 4) * 32 + lane;

    for (int kk = 0; kk < Dm; kk += 32) {
        v16bf a0 = concat8(aPtr[0], aPtr[1]);
        v16bf a1 = concat8(aPtr[aOff1], aPtr[aOff1 + 1]);
        v16bf b0 = bPtr[0], b1 = bPtr[32], b2 = bPtr[64], b3 = bPtr[96];

        acc0[0] = wmma_bf16(a0, b0, acc0[0]);
        acc1[0] = wmma_bf16(a1, b0, acc1[0]);
        acc0[1] = wmma_bf16(a0, b1, acc0[1]);
        acc1[1] = wmma_bf16(a1, b1, acc1[1]);
        acc0[2] = wmma_bf16(a0, b2, acc0[2]);
        acc1[2] = wmma_bf16(a1, b2, acc1[2]);
        acc0[3] = wmma_bf16(a0, b3, acc0[3]);
        acc1[3] = wmma_bf16(a1, b3, acc1[3]);

        aPtr += 4;        // 32 bf16
        bPtr += bStep;
    }

    // ---- store: C/D layout VGPR r -> row r+8*half, lane&15 -> col ----
    if (outF) {
#pragma unroll
        for (int t = 0; t < 4; ++t)
#pragma unroll
            for (int r = 0; r < 8; ++r) {
                int row = rowBase + r + 8 * half;
                int col = colBase + t * 16 + l15;
                float bc = bias[col];
                outF[(size_t)row * Dm + col]        = acc0[t][r] + bc;
                outF[(size_t)(row + 16) * Dm + col] = acc1[t][r] + bc;
            }
    } else {
#pragma unroll
        for (int t = 0; t < 4; ++t)
#pragma unroll
            for (int r = 0; r < 8; ++r) {
                int row = rowBase + r + 8 * half;
                int col = colBase + t * 16 + l15;
                float bc = bias[col];
                outBf[(size_t)row * Dm + col]        = f_to_bfbits(acc0[t][r] + bc);
                outBf[(size_t)(row + 16) * Dm + col] = f_to_bfbits(acc1[t][r] + bc);
            }
    }
}

// ------------- fused causal scores + top-8 + softmax + gather --------------
// grid = (Nseq/16, Bsz), blockDim = 32 (one wave per 16 query rows).
// Each pass computes 4 key tiles sharing one q fragment per k-step.
__global__ void attn_topk_kernel(const __bf16* __restrict__ qbf,
                                 const __bf16* __restrict__ kbf,
                                 const unsigned short* __restrict__ vbf,
                                 unsigned short* __restrict__ attnOut) {
    __shared__ float sS[4 * 256];   // four 16x16 score tiles
    __shared__ float sW[16][TOPK];  // softmax weights
    __shared__ int   sI[16][TOPK];  // gather indices

    const int lane = threadIdx.x & 31;
    const int half = lane >> 4;
    const int l15  = lane & 15;
    const int b    = blockIdx.y;
    const int nt   = blockIdx.x;

    const __bf16* Aq = qbf + ((size_t)b * Nseq + nt * 16 + l15) * Dm + 16 * half;
    const __bf16* Kb = kbf + ((size_t)b * Nseq + l15) * Dm + 16 * half;

    float tv[TOPK];
    int   ti[TOPK];
#pragma unroll
    for (int j = 0; j < TOPK; ++j) { tv[j] = -__builtin_inff(); ti[j] = -1; }

    const float scale = 0.03125f;  // 1/sqrt(1024)

    for (int mtBase = 0; mtBase <= nt; mtBase += 4) {
        // clamped key-row pointers for the 4 sub-tiles (overshoot masked later)
        int mt1 = mtBase + 1 > nt ? nt : mtBase + 1;
        int mt2 = mtBase + 2 > nt ? nt : mtBase + 2;
        int mt3 = mtBase + 3 > nt ? nt : mtBase + 3;
        const v16bf* bk0 = (const v16bf*)(Kb + (size_t)mtBase * 16 * Dm);
        const v16bf* bk1 = (const v16bf*)(Kb + (size_t)mt1 * 16 * Dm);
        const v16bf* bk2 = (const v16bf*)(Kb + (size_t)mt2 * 16 * Dm);
        const v16bf* bk3 = (const v16bf*)(Kb + (size_t)mt3 * 16 * Dm);

        const v8bf* aq = (const v8bf*)Aq;
        v8f s0 = {}, s1 = {}, s2 = {}, s3 = {};
        for (int kk = 0; kk < Dm; kk += 32) {
            v16bf a = concat8(aq[0], aq[1]);
            aq += 4;
            s0 = wmma_bf16(a, bk0[0], s0); bk0 += 2;
            s1 = wmma_bf16(a, bk1[0], s1); bk1 += 2;
            s2 = wmma_bf16(a, bk2[0], s2); bk2 += 2;
            s3 = wmma_bf16(a, bk3[0], s3); bk3 += 2;
        }

        // causal mask + scale, spill 4 tiles to LDS
#pragma unroll
        for (int r = 0; r < 8; ++r) {
            const int n    = nt * 16 + r + 8 * half;
            const int rowo = (r + 8 * half) * 16 + l15;
            const float NI = -__builtin_inff();
            int m0 = mtBase * 16 + l15;
            sS[0 * 256 + rowo] = (m0       <= n)                     ? s0[r] * scale : NI;
            sS[1 * 256 + rowo] = (m0 + 16  <= n && mtBase + 1 <= nt) ? s1[r] * scale : NI;
            sS[2 * 256 + rowo] = (m0 + 32  <= n && mtBase + 2 <= nt) ? s2[r] * scale : NI;
            sS[3 * 256 + rowo] = (m0 + 48  <= n && mtBase + 3 <= nt) ? s3[r] * scale : NI;
        }
        __syncthreads();
        // lanes 0..15: scan 64 candidates, maintain per-row sorted top-8
        if (lane < 16) {
#pragma unroll
            for (int t = 0; t < 4; ++t) {
#pragma unroll
                for (int j = 0; j < 16; ++j) {
                    float val = sS[t * 256 + lane * 16 + j];
                    if (val > tv[TOPK - 1]) {
                        tv[TOPK - 1] = val;
                        ti[TOPK - 1] = (mtBase + t) * 16 + j;
#pragma unroll
                        for (int p = TOPK - 1; p >= 1; --p) {
                            if (tv[p] > tv[p - 1]) {
                                float tf = tv[p]; tv[p] = tv[p - 1]; tv[p - 1] = tf;
                                int   t2 = ti[p]; ti[p] = ti[p - 1]; ti[p - 1] = t2;
                            }
                        }
                    }
                }
            }
        }
        __syncthreads();
    }

    // sparse softmax over the 8 survivors (tv sorted descending; tv[0] finite:
    // the diagonal m==n is always a valid candidate)
    if (lane < 16) {
        float mx = tv[0];
        float w[TOPK], sum = 0.f;
#pragma unroll
        for (int j = 0; j < TOPK; ++j) { w[j] = __expf(tv[j] - mx); sum += w[j]; }
        float inv = 1.f / sum;
#pragma unroll
        for (int j = 0; j < TOPK; ++j) {
            sW[lane][j] = w[j] * inv;
            sI[lane][j] = (ti[j] < 0) ? 0 : ti[j];  // weight is 0 when idx invalid
        }
    }
    __syncthreads();

    // out[n,:] = sum_j w_j * v[idx_j,:]  (8-way gather, v is L2-resident)
    for (int r = 0; r < 16; ++r) {
        float w[TOPK];
        const v4u* vp[TOPK];
#pragma unroll
        for (int j = 0; j < TOPK; ++j) {
            w[j]  = sW[r][j];
            vp[j] = (const v4u*)(vbf + ((size_t)b * Nseq + sI[r][j]) * Dm);
        }
        v4u* op = (v4u*)(attnOut + ((size_t)b * Nseq + nt * 16 + r) * Dm);
        for (int c = lane; c < Dm / 8; c += 32) {  // 8 bf16 per v4u
            float acc[8] = {0.f, 0.f, 0.f, 0.f, 0.f, 0.f, 0.f, 0.f};
#pragma unroll
            for (int j = 0; j < TOPK; ++j) {
                v4u u = vp[j][c];
#pragma unroll
                for (int e = 0; e < 4; ++e) {
                    acc[2 * e]     += w[j] * bfbits_to_f(u[e] & 0xffffu);
                    acc[2 * e + 1] += w[j] * bfbits_to_f(u[e] >> 16);
                }
            }
            v4u o;
#pragma unroll
            for (int e = 0; e < 4; ++e)
                o[e] = (unsigned int)f_to_bfbits(acc[2 * e]) |
                       ((unsigned int)f_to_bfbits(acc[2 * e + 1]) << 16);
            op[c] = o;
        }
    }
}

// ------------------------------- launcher ----------------------------------
extern "C" void kernel_launch(void* const* d_in, const int* in_sizes, int n_in,
                              void* d_out, int out_size, void* d_ws, size_t ws_size,
                              hipStream_t stream) {
    const float* X  = (const float*)d_in[0];
    const float* Wq = (const float*)d_in[1];
    const float* bq = (const float*)d_in[2];
    const float* Wk = (const float*)d_in[3];
    const float* bk = (const float*)d_in[4];
    const float* Wv = (const float*)d_in[5];
    const float* bv = (const float*)d_in[6];
    const float* Wo = (const float*)d_in[7];
    const float* bo = (const float*)d_in[8];
    float* out = (float*)d_out;

    const size_t MD = (size_t)Bsz * Nseq * Dm;  // 16,777,216
    const size_t DD = (size_t)Dm * Dm;          //  1,048,576

    char* ws = (char*)d_ws;
    unsigned short* Xbf    = (unsigned short*)ws; ws += MD * 2;
    unsigned int*   Wqpk   = (unsigned int*)ws;   ws += DD * 2;  // DD/2 uints
    unsigned int*   Wkpk   = (unsigned int*)ws;   ws += DD * 2;
    unsigned int*   Wvpk   = (unsigned int*)ws;   ws += DD * 2;
    unsigned int*   Wopk   = (unsigned int*)ws;   ws += DD * 2;
    unsigned short* qbf    = (unsigned short*)ws; ws += MD * 2;
    unsigned short* kbf    = (unsigned short*)ws; ws += MD * 2;
    unsigned short* vbf    = (unsigned short*)ws; ws += MD * 2;
    unsigned short* attnbf = (unsigned short*)ws; ws += MD * 2;

    // 1) convert activations, pack weights into WMMA fragment order
    cvt_f32_to_bf16<<<2048, 256, 0, stream>>>((const float2*)X,
                                              (unsigned int*)Xbf, (int)(MD / 2));
    pack_weight_bf16<<<512, 256, 0, stream>>>(Wq, Wqpk);
    pack_weight_bf16<<<512, 256, 0, stream>>>(Wk, Wkpk);
    pack_weight_bf16<<<512, 256, 0, stream>>>(Wv, Wvpk);
    pack_weight_bf16<<<512, 256, 0, stream>>>(Wo, Wopk);

    // 2) q/k/v projections (WMMA, bf16 out)
    const int M = Bsz * Nseq;
    dim3 ggrid(Dm / 256, M / 32);
    wmma_gemm_bf16<<<ggrid, 128, 0, stream>>>((const __bf16*)Xbf, (const v16bf*)Wqpk,
                                              bq, qbf, nullptr);
    wmma_gemm_bf16<<<ggrid, 128, 0, stream>>>((const __bf16*)Xbf, (const v16bf*)Wkpk,
                                              bk, kbf, nullptr);
    wmma_gemm_bf16<<<ggrid, 128, 0, stream>>>((const __bf16*)Xbf, (const v16bf*)Wvpk,
                                              bv, vbf, nullptr);

    // 3) causal scores + top-8 sparse softmax + gathered attn*v (WMMA scores)
    attn_topk_kernel<<<dim3(Nseq / 16, Bsz), 32, 0, stream>>>(
        (const __bf16*)qbf, (const __bf16*)kbf, vbf, attnbf);

    // 4) output projection (WMMA, f32 out)
    wmma_gemm_bf16<<<ggrid, 128, 0, stream>>>((const __bf16*)attnbf, (const v16bf*)Wopk,
                                              bo, nullptr, out);
}
